// BoostingMoE_8100308320513
// MI455X (gfx1250) — matrix operand order repaired and checked by reading, not compiled, not run on hardware
//
#include <hip/hip_runtime.h>

typedef unsigned short u16;
typedef __attribute__((ext_vector_type(16))) __bf16 v16bf;
typedef __attribute__((ext_vector_type(8)))  __bf16 v8bf;
typedef __attribute__((ext_vector_type(8)))  float  v8f;

union FragAB { v16bf v; v8bf h[2]; };

__device__ __forceinline__ u16 f2bf(float x) {
  unsigned int u = __float_as_uint(x);
  unsigned int r = (u + 0x7FFFu + ((u >> 16) & 1u)) >> 16;
  return (u16)r;
}

// ---------------------------------------------------------------------------
// Elementwise: x -> x_bf16, boost_f32 = x, boost_bf16 = bf16(x). 4 floats/thread.
// ---------------------------------------------------------------------------
__global__ __launch_bounds__(256) void conv_x_kernel(
    const float* __restrict__ x, u16* __restrict__ x_bf,
    float* __restrict__ boost_f, u16* __restrict__ boost_bf) {
  long i = ((long)blockIdx.x * 256 + threadIdx.x) * 4;
  float4 v = *(const float4*)(x + i);
  *(float4*)(boost_f + i) = v;
  ushort4 b;
  b.x = f2bf(v.x); b.y = f2bf(v.y); b.z = f2bf(v.z); b.w = f2bf(v.w);
  *(ushort4*)(x_bf + i) = b;
  *(ushort4*)(boost_bf + i) = b;
}

// ---------------------------------------------------------------------------
// Transpose-convert: W[k][n] f32 row-major -> Wt[n][k] bf16 (K-contiguous).
// 32x32 tiles through LDS. blockIdx.z = expert.
// ---------------------------------------------------------------------------
__global__ __launch_bounds__(256) void transpose_to_bf16(
    const float* __restrict__ W, u16* __restrict__ Wt,
    int K, int N, long sIn, long sOut) {
  __shared__ u16 t[32][33];
  W  += (long)blockIdx.z * sIn;
  Wt += (long)blockIdx.z * sOut;
  const int lx = threadIdx.x & 31, ly = threadIdx.x >> 5;  // ly: 0..7
  const int kt = blockIdx.y * 32, nt = blockIdx.x * 32;
#pragma unroll
  for (int q = 0; q < 4; ++q) {
    int kk = ly + q * 8;
    t[kk][lx] = f2bf(W[(long)(kt + kk) * N + nt + lx]);
  }
  __syncthreads();
#pragma unroll
  for (int q = 0; q < 4; ++q) {
    int nn = ly + q * 8;
    Wt[(long)(nt + nn) * K + kt + lx] = t[lx][nn];
  }
}

// ---------------------------------------------------------------------------
// BF16 WMMA GEMM: C[M,N] = A[M,K] * Bt[N,K]^T (+bias).
// Block = 256 threads (8 waves), tile 128x128, K-step 32, DOUBLE-BUFFERED LDS:
// one barrier per k-step; next tile's global loads overlap current WMMAs.
// Wave grid 4x2; each wave owns 32x64 = 2x4 WMMA 16x16 accumulators.
// MODE 0: out_bf16 = bf16(relu(acc+bias))
// MODE 1: eo = acc+bias; out_bf16 = bf16(eo); boost_f += 0.5*eo; boost_bf = bf16(boost_f)
// MODE 2: out_f32 = acc+bias
// blockIdx.z = expert (strides select slices; 0 = shared).
// ---------------------------------------------------------------------------
#define LDSW 40        // 32 + 8 pad (u16 units); row stride 80B, 16B aligned
#define TILEW (128 * LDSW)

template <int MODE>
__global__ __launch_bounds__(256) void gemm_wmma_bf16(
    const u16* __restrict__ A, long sAe,
    const u16* __restrict__ Bt, long sBe,
    const float* __restrict__ bias, long sBiasE,
    u16* __restrict__ oBf, long sObfE,
    float* __restrict__ oF, long sOfE,
    float* __restrict__ boostF, u16* __restrict__ boostBf,
    int M, int N, int K) {
  __shared__ u16 lA[2][TILEW];
  __shared__ u16 lB[2][TILEW];

  const long e = blockIdx.z;
  A    += e * sAe;
  Bt   += e * sBe;
  bias += e * sBiasE;
  oBf  += e * sObfE;
  oF   += e * sOfE;

  const int tid  = threadIdx.x;
  const int lane = tid & 31;
  const int wave = tid >> 5;
  const int wm   = wave >> 1;   // 0..3 -> M sub-tile
  const int wn   = wave & 1;    // 0..1 -> N sub-tile
  const int g    = lane >> 4;   // half-wave
  const int l16  = lane & 15;

  const int tm0 = blockIdx.y * 128;
  const int tn0 = blockIdx.x * 128;

  // Per-thread staging assignment: two 16B chunks of A and two of B per step.
  const int row0 = tid >> 2,        kc0 = (tid & 3) << 3;
  const int row1 = (tid + 256) >> 2, kc1 = ((tid + 256) & 3) << 3;
  const u16* gA0 = &A[(long)(tm0 + row0) * K + kc0];
  const u16* gA1 = &A[(long)(tm0 + row1) * K + kc1];
  const u16* gB0 = &Bt[(long)(tn0 + row0) * K + kc0];
  const u16* gB1 = &Bt[(long)(tn0 + row1) * K + kc1];
  u16* sA0 = &lA[0][row0 * LDSW + kc0];
  u16* sA1 = &lA[0][row1 * LDSW + kc1];
  u16* sB0 = &lB[0][row0 * LDSW + kc0];
  u16* sB1 = &lB[0][row1 * LDSW + kc1];

  v8f acc[2][4];
#pragma unroll
  for (int mi = 0; mi < 2; ++mi)
#pragma unroll
    for (int ni = 0; ni < 4; ++ni) {
      v8f z = {};
      acc[mi][ni] = z;
    }

  const int nsteps = K >> 5;
  uint4 ra0, ra1, rb0, rb1;

  // Prologue: stage tile 0 into buffer 0.
  ra0 = *(const uint4*)(gA0);
  ra1 = *(const uint4*)(gA1);
  rb0 = *(const uint4*)(gB0);
  rb1 = *(const uint4*)(gB1);
  *(uint4*)(sA0) = ra0;
  *(uint4*)(sA1) = ra1;
  *(uint4*)(sB0) = rb0;
  *(uint4*)(sB1) = rb1;
  __syncthreads();

  int buf = 0;
  for (int s = 0; s < nsteps; ++s) {
    // Issue next tile's global loads early (overlap with WMMA below).
    if (s + 1 < nsteps) {
      const long ko = (long)(s + 1) << 5;
      ra0 = *(const uint4*)(gA0 + ko);
      ra1 = *(const uint4*)(gA1 + ko);
      rb0 = *(const uint4*)(gB0 + ko);
      rb1 = *(const uint4*)(gB1 + ko);
    }

    // Compute from current buffer.
    const u16* bA = &lA[buf][0];
    const u16* bB = &lB[buf][0];
    FragAB fa[2];
#pragma unroll
    for (int mi = 0; mi < 2; ++mi) {
      // A fragment (16-bit 16x32 layout): lane row = l16, K chunks g*8, 16+g*8.
      const u16* p = bA + (wm * 32 + mi * 16 + l16) * LDSW;
      fa[mi].h[0] = *(const v8bf*)(p + g * 8);
      fa[mi].h[1] = *(const v8bf*)(p + 16 + g * 8);
    }
    FragAB fb[4];
#pragma unroll
    for (int ni = 0; ni < 4; ++ni) {
      // B fragment: lane col = l16, K chunks g*8, 16+g*8.
      const u16* p = bB + (wn * 64 + ni * 16 + l16) * LDSW;
      fb[ni].h[0] = *(const v8bf*)(p + g * 8);
      fb[ni].h[1] = *(const v8bf*)(p + 16 + g * 8);
    }
#pragma unroll
    for (int mi = 0; mi < 2; ++mi)
#pragma unroll
      for (int ni = 0; ni < 4; ++ni)
        acc[mi][ni] = __builtin_amdgcn_wmma_f32_16x16x32_bf16(
            false, fa[mi].v, false, fb[ni].v, (short)0, acc[mi][ni], false,
            false);

    // Store next tile into the other buffer (no conflict with reads above).
    if (s + 1 < nsteps) {
      const int o = (buf ^ 1) * TILEW;
      *(uint4*)(sA0 + o) = ra0;
      *(uint4*)(sA1 + o) = ra1;
      *(uint4*)(sB0 + o) = rb0;
      *(uint4*)(sB1 + o) = rb1;
    }
    __syncthreads();
    buf ^= 1;
  }

  // Epilogue. C/D layout: VGPR r -> row r + 8*g; col = l16.
#pragma unroll
  for (int mi = 0; mi < 2; ++mi) {
#pragma unroll
    for (int ni = 0; ni < 4; ++ni) {
      const int col = tn0 + wn * 64 + ni * 16 + l16;
      const float b = bias[col];
#pragma unroll
      for (int r = 0; r < 8; ++r) {
        const int row = tm0 + wm * 32 + mi * 16 + g * 8 + r;
        const long o = (long)row * N + col;
        const float t = acc[mi][ni][r] + b;
        if (MODE == 0) {
          oBf[o] = f2bf(t > 0.f ? t : 0.f);
        } else if (MODE == 1) {
          oBf[o] = f2bf(t);
          const float nb = boostF[o] + 0.5f * t;  // ALPHA = 0.5
          boostF[o] = nb;
          boostBf[o] = f2bf(nb);
        } else {
          oF[o] = t;
        }
      }
    }
  }
}

// ---------------------------------------------------------------------------
// Per-token: scores = q.k_e/sqrt(H), softmax over 8 experts, fused = sum w*v_e,
// y = x + fused, LayerNorm. One wave32 per token, 8 waves per block.
// ---------------------------------------------------------------------------
#define HDIM 1024
#define NEXP 8

__global__ __launch_bounds__(256) void fuse_softmax_ln(
    const float* __restrict__ x, const float* __restrict__ q,
    const float* __restrict__ k, const float* __restrict__ v,
    const float* __restrict__ ln_g, const float* __restrict__ ln_b,
    float* __restrict__ out, long tokStride) {
  const int lane = threadIdx.x & 31;
  const int wave = threadIdx.x >> 5;
  const long token = (long)blockIdx.x * 8 + wave;
  const long base = token * HDIM;

  float s[NEXP];
#pragma unroll
  for (int e = 0; e < NEXP; ++e) s[e] = 0.f;

#pragma unroll
  for (int it = 0; it < HDIM / 32; ++it) {
    const int h = lane + it * 32;
    const float qv = q[base + h];
#pragma unroll
    for (int e = 0; e < NEXP; ++e)
      s[e] += qv * k[(long)e * tokStride + base + h];
  }
#pragma unroll
  for (int e = 0; e < NEXP; ++e)
#pragma unroll
    for (int off = 16; off > 0; off >>= 1) s[e] += __shfl_xor(s[e], off, 32);

  const float scale = 0.03125f;  // 1/sqrt(1024)
  float mx = -3.4e38f;
#pragma unroll
  for (int e = 0; e < NEXP; ++e) {
    s[e] *= scale;
    mx = fmaxf(mx, s[e]);
  }
  float sum = 0.f;
#pragma unroll
  for (int e = 0; e < NEXP; ++e) {
    s[e] = __expf(s[e] - mx);
    sum += s[e];
  }
  const float isum = 1.f / sum;
#pragma unroll
  for (int e = 0; e < NEXP; ++e) s[e] *= isum;

  float yreg[HDIM / 32];
  float a1 = 0.f, a2 = 0.f;
#pragma unroll
  for (int it = 0; it < HDIM / 32; ++it) {
    const int h = lane + it * 32;
    float f = 0.f;
#pragma unroll
    for (int e = 0; e < NEXP; ++e)
      f += s[e] * v[(long)e * tokStride + base + h];
    const float yv = x[base + h] + f;
    yreg[it] = yv;
    a1 += yv;
    a2 += yv * yv;
  }
#pragma unroll
  for (int off = 16; off > 0; off >>= 1) {
    a1 += __shfl_xor(a1, off, 32);
    a2 += __shfl_xor(a2, off, 32);
  }
  const float mu = a1 * (1.f / HDIM);
  const float var = a2 * (1.f / HDIM) - mu * mu;
  const float rinv = rsqrtf(var + 1e-5f);
#pragma unroll
  for (int it = 0; it < HDIM / 32; ++it) {
    const int h = lane + it * 32;
    out[base + h] = (yreg[it] - mu) * rinv * ln_g[h] + ln_b[h];
  }
}

// ---------------------------------------------------------------------------
extern "C" void kernel_launch(void* const* d_in, const int* in_sizes, int n_in,
                              void* d_out, int out_size, void* d_ws,
                              size_t ws_size, hipStream_t stream) {
  const long TOK = 16384, H = 1024, DH = 4096, E = 8;

  const float* x    = (const float*)d_in[0];
  const float* We1  = (const float*)d_in[1];
  const float* be1  = (const float*)d_in[2];
  const float* We2  = (const float*)d_in[3];
  const float* be2  = (const float*)d_in[4];
  const float* Wq   = (const float*)d_in[5];
  const float* bq   = (const float*)d_in[6];
  const float* Wk   = (const float*)d_in[7];
  const float* bk   = (const float*)d_in[8];
  const float* Wv   = (const float*)d_in[9];
  const float* bv   = (const float*)d_in[10];
  const float* ln_g = (const float*)d_in[11];
  const float* ln_b = (const float*)d_in[12];
  float* out = (float*)d_out;

  char* w = (char*)d_ws;
  size_t off = 0;
  auto alloc = [&](size_t bytes) -> void* {
    void* p = w + off;
    off = (off + bytes + 255) & ~(size_t)255;
    return p;
  };
  u16*   x_bf     = (u16*)alloc(TOK * H * 2);
  float* boost_f  = (float*)alloc(TOK * H * 4);
  u16*   boost_bf = (u16*)alloc(TOK * H * 2);
  u16*   h1_bf    = (u16*)alloc(TOK * DH * 2);
  u16*   eo_bf    = (u16*)alloc(E * TOK * H * 2);
  float* q_f      = (float*)alloc(TOK * H * 4);
  float* k_f      = (float*)alloc(E * TOK * H * 4);
  float* v_f      = (float*)alloc(E * TOK * H * 4);
  u16*   We1t     = (u16*)alloc(E * H * DH * 2);
  u16*   We2t     = (u16*)alloc(E * DH * H * 2);
  u16*   Wqt      = (u16*)alloc(H * H * 2);
  u16*   Wkt      = (u16*)alloc(H * H * 2);
  u16*   Wvt      = (u16*)alloc(H * H * 2);

  // 1) x -> bf16, boost init
  conv_x_kernel<<<dim3(TOK * H / (256 * 4)), 256, 0, stream>>>(x, x_bf, boost_f,
                                                               boost_bf);
  // 2) weights -> bf16, transposed [n][k]
  transpose_to_bf16<<<dim3(DH / 32, H / 32, E), 256, 0, stream>>>(
      We1, We1t, H, DH, H * DH, H * DH);
  transpose_to_bf16<<<dim3(H / 32, DH / 32, E), 256, 0, stream>>>(
      We2, We2t, DH, H, DH * H, DH * H);
  transpose_to_bf16<<<dim3(H / 32, H / 32, 1), 256, 0, stream>>>(Wq, Wqt, H, H,
                                                                 0, 0);
  transpose_to_bf16<<<dim3(H / 32, H / 32, 1), 256, 0, stream>>>(Wk, Wkt, H, H,
                                                                 0, 0);
  transpose_to_bf16<<<dim3(H / 32, H / 32, 1), 256, 0, stream>>>(Wv, Wvt, H, H,
                                                                 0, 0);

  // 3) Sequential boosting chain: 8 experts (n = current_expert_idx+1 = E)
  for (int i = 0; i < (int)E; ++i) {
    gemm_wmma_bf16<0><<<dim3(DH / 128, TOK / 128, 1), 256, 0, stream>>>(
        boost_bf, 0, We1t + (long)i * H * DH, 0, be1 + (long)i * DH, 0, h1_bf,
        0, nullptr, 0, nullptr, nullptr, (int)TOK, (int)DH, (int)H);
    gemm_wmma_bf16<1><<<dim3(H / 128, TOK / 128, 1), 256, 0, stream>>>(
        h1_bf, 0, We2t + (long)i * DH * H, 0, be2 + (long)i * H, 0,
        eo_bf + (long)i * TOK * H, 0, nullptr, 0, boost_f, boost_bf, (int)TOK,
        (int)H, (int)DH);
  }

  // 4) Q projection (from x), K/V projections (from all 8 expert outputs)
  gemm_wmma_bf16<2><<<dim3(H / 128, TOK / 128, 1), 256, 0, stream>>>(
      x_bf, 0, Wqt, 0, bq, 0, nullptr, 0, q_f, 0, nullptr, nullptr, (int)TOK,
      (int)H, (int)H);
  gemm_wmma_bf16<2><<<dim3(H / 128, TOK / 128, E), 256, 0, stream>>>(
      eo_bf, TOK * H, Wkt, 0, bk, 0, nullptr, 0, k_f, TOK * H, nullptr,
      nullptr, (int)TOK, (int)H, (int)H);
  gemm_wmma_bf16<2><<<dim3(H / 128, TOK / 128, E), 256, 0, stream>>>(
      eo_bf, TOK * H, Wvt, 0, bv, 0, nullptr, 0, v_f, TOK * H, nullptr,
      nullptr, (int)TOK, (int)H, (int)H);

  // 5) Softmax fuse + residual + LayerNorm
  fuse_softmax_ln<<<dim3(TOK / 8), 256, 0, stream>>>(x, q_f, k_f, v_f, ln_g,
                                                     ln_b, out, TOK * H);
}
